// HierarchicalGAT_43276090474717
// MI455X (gfx1250) — compile-verified, matrix operand order checked by reading, not run on hardware
//
#include <hip/hip_runtime.h>
#include <hip/hip_bf16.h>

// ---------------------------------------------------------------------------
// Problem dims (fixed by the reference)
// ---------------------------------------------------------------------------
#define N_FINE   32768
#define N_COARSE 2048
#define BSZ      32
#define IN_F     1024
#define HID      2048
#define OUT_P    128
#define ATT      64
#define H1       8
#define H2       4
#define LLM      4096
#define CD       512           // ATT*H1 == H2*OUT_P
#define OUT_ROWS 1089          // 1 + 64 + 1024

typedef __attribute__((ext_vector_type(16))) __bf16 v16bf;
typedef __attribute__((ext_vector_type(8)))  float  v8f;

union Frag { v16bf v; uint4 q[2]; };

// ---------------------------------------------------------------------------
// WMMA GEMM: C[M,N] = act(A[M,K] (bf16, row-major) @ Bt[N,K]^T (bf16) + bias)
// Each wave computes a 32x64 tile (2 A-frags x 4 B-frags = 8 WMMAs per
// k-step); block = 4 waves covering 128 rows. All M in this net are
// multiples of 32, so tiles are always full and EXEC stays all-1s.
// Optional f32 output (non-temporal, with batched "1089-row" mapping) and
// optional bf16 output for chaining into the next GEMM.
// ---------------------------------------------------------------------------
__global__ __launch_bounds__(128)
void gemm_wmma_bf16_kernel(const __hip_bfloat16* __restrict__ A,
                           const __hip_bfloat16* __restrict__ Bt,
                           const float* __restrict__ bias,
                           float* __restrict__ outF,
                           __hip_bfloat16* __restrict__ outB,
                           int M, int N, int K,
                           int group_rows, int group_offset,
                           long long out_batch_stride, int out_row_stride,
                           int relu)
{
    const int wave = threadIdx.x >> 5;
    const int lane = threadIdx.x & 31;
    const int half = lane >> 4;
    const int l16  = lane & 15;

    const int m0 = (blockIdx.y * 4 + wave) * 32;
    const int n0 = blockIdx.x * 64;
    if (m0 >= M) return;                       // wave-uniform: EXEC stays all-1s

    const __hip_bfloat16* arow0 = A + (size_t)(m0 + l16) * K;
    const __hip_bfloat16* arow1 = A + (size_t)(m0 + 16 + l16) * K;
    const __hip_bfloat16* brow[4];
    #pragma unroll
    for (int t = 0; t < 4; ++t)
        brow[t] = Bt + (size_t)(n0 + t * 16 + l16) * K;

    v8f acc[2][4];
    #pragma unroll
    for (int i = 0; i < 2; ++i)
        #pragma unroll
        for (int t = 0; t < 4; ++t)
            acc[i][t] = (v8f){};

    const int o0 = 8 * half;          // lanes 0-15: K=0..7,16..23 ; lanes 16-31: K=8..15,24..31
    const int o1 = 16 + 8 * half;

    for (int kk = 0; kk < K; kk += 32) {
        __builtin_prefetch((const void*)(arow0 + kk + 64), 0, 0);
        __builtin_prefetch((const void*)(arow1 + kk + 64), 0, 0);

        Frag a0, a1;
        a0.q[0] = *(const uint4*)(arow0 + kk + o0);
        a0.q[1] = *(const uint4*)(arow0 + kk + o1);
        a1.q[0] = *(const uint4*)(arow1 + kk + o0);
        a1.q[1] = *(const uint4*)(arow1 + kk + o1);

        #pragma unroll
        for (int t = 0; t < 4; ++t) {
            Frag b;
            b.q[0] = *(const uint4*)(brow[t] + kk + o0);
            b.q[1] = *(const uint4*)(brow[t] + kk + o1);
            acc[0][t] = __builtin_amdgcn_wmma_f32_16x16x32_bf16(false, a0.v, false, b.v,
                                                               (short)0, acc[0][t], false, false);
            acc[1][t] = __builtin_amdgcn_wmma_f32_16x16x32_bf16(false, a1.v, false, b.v,
                                                               (short)0, acc[1][t], false, false);
        }
    }

    #pragma unroll
    for (int i = 0; i < 2; ++i) {
        #pragma unroll
        for (int t = 0; t < 4; ++t) {
            const int col = n0 + t * 16 + l16;
            const float bv = bias ? bias[col] : 0.0f;
            #pragma unroll
            for (int r = 0; r < 8; ++r) {
                const int row = m0 + 16 * i + r + 8 * half;  // C layout: M = r + 8*(lane/16)
                float v = acc[i][t][r] + bv;
                if (relu) v = fmaxf(v, 0.0f);
                if (outF) {
                    size_t o;
                    if (group_rows > 0) {
                        const int bb = row / group_rows;
                        const int j  = row % group_rows;
                        o = (size_t)bb * (size_t)out_batch_stride
                          + (size_t)(group_offset + j) * (size_t)out_row_stride + col;
                    } else {
                        o = (size_t)row * (size_t)out_row_stride + col;
                    }
                    __builtin_nontemporal_store(v, outF + o);   // write-once output
                }
                if (outB) outB[(size_t)row * N + col] = __float2bfloat16(v);
            }
        }
    }
}

// ---------------------------------------------------------------------------
// LayerNorm over rows; optional ReLU; optional f32 and/or bf16 outputs.
// ---------------------------------------------------------------------------
__global__ __launch_bounds__(256)
void ln_kernel(const float* __restrict__ x, const float* __restrict__ g,
               const float* __restrict__ b, float* __restrict__ of,
               __hip_bfloat16* __restrict__ ob, int D, int relu)
{
    __shared__ float ssum[256], ssq[256];
    const int row = blockIdx.x;
    const float* xr = x + (size_t)row * D;
    float sum = 0.f, sq = 0.f;
    for (int i = threadIdx.x; i < D; i += blockDim.x) {
        float v = xr[i]; sum += v; sq += v * v;
    }
    ssum[threadIdx.x] = sum; ssq[threadIdx.x] = sq;
    __syncthreads();
    for (int s = 128; s > 0; s >>= 1) {
        if (threadIdx.x < s) {
            ssum[threadIdx.x] += ssum[threadIdx.x + s];
            ssq[threadIdx.x]  += ssq[threadIdx.x + s];
        }
        __syncthreads();
    }
    const float mu   = ssum[0] / (float)D;
    const float var  = ssq[0] / (float)D - mu * mu;
    const float rstd = rsqrtf(var + 1e-5f);
    for (int i = threadIdx.x; i < D; i += blockDim.x) {
        float v = (xr[i] - mu) * rstd * g[i] + b[i];
        if (relu) v = fmaxf(v, 0.0f);
        if (of) of[(size_t)row * D + i] = v;
        if (ob) ob[(size_t)row * D + i] = __float2bfloat16(v);
    }
}

// ---------------------------------------------------------------------------
// Cast f32 W[K,N] -> bf16 Wt[N,K] (transposed for WMMA B-fragment loads)
// ---------------------------------------------------------------------------
__global__ __launch_bounds__(256)
void castT_kernel(const float* __restrict__ W, __hip_bfloat16* __restrict__ Wt,
                  int K, int N)
{
    int idx = blockIdx.x * blockDim.x + threadIdx.x;
    if (idx >= K * N) return;
    int k = idx / N, n = idx % N;
    Wt[(size_t)n * K + k] = __float2bfloat16(W[idx]);
}

// ---------------------------------------------------------------------------
// Per-node attention scores: s[n,h] = sum_c h[n,h*C+c] * a[h*C+c]
// ---------------------------------------------------------------------------
__global__ __launch_bounds__(256)
void scores_kernel(const float* __restrict__ h, const float* __restrict__ a,
                   float* __restrict__ s, int N, int H, int C)
{
    int idx = blockIdx.x * blockDim.x + threadIdx.x;
    if (idx >= N * H) return;
    int n = idx / H, hh = idx % H;
    const float* hp = h + (size_t)n * H * C + hh * C;
    const float* ap = a + (size_t)hh * C;
    float acc = 0.f;
    for (int c = 0; c < C; ++c) acc += hp[c] * ap[c];
    s[idx] = acc;
}

__global__ __launch_bounds__(256)
void fill_u32_kernel(unsigned* __restrict__ p, unsigned v, int n)
{
    int idx = blockIdx.x * blockDim.x + threadIdx.x;
    if (idx < n) p[idx] = v;
}

// Monotone float<->uint encoding for atomicMax on signed floats
__device__ __forceinline__ unsigned fenc(float x) {
    unsigned u = __float_as_uint(x);
    return (u & 0x80000000u) ? ~u : (u | 0x80000000u);
}
__device__ __forceinline__ float fdec(unsigned k) {
    return (k & 0x80000000u) ? __uint_as_float(k & 0x7fffffffu) : __uint_as_float(~k);
}

// Pass 1: e = leaky_relu(ssrc[s]+sdst[d]); segment-max via atomicMax on keys
__global__ __launch_bounds__(256)
void edge_max_kernel(const int* __restrict__ src, const int* __restrict__ dst,
                     int E, int nself, int H,
                     const float* __restrict__ ssrc, const float* __restrict__ sdst,
                     float* __restrict__ ebuf, unsigned* __restrict__ mkey)
{
    int idx = blockIdx.x * blockDim.x + threadIdx.x;
    int ET = E + nself;
    if (idx >= ET * H) return;
    int e = idx / H, h = idx % H;
    int s_ = (e < E) ? src[e] : (e - E);
    int d_ = (e < E) ? dst[e] : (e - E);
    float v = ssrc[s_ * H + h] + sdst[d_ * H + h];
    v = (v > 0.f) ? v : 0.2f * v;
    ebuf[idx] = v;
    atomicMax(&mkey[d_ * H + h], fenc(v));
}

// Pass 2: ex = exp(e - m[d]); segment-sum via atomicAdd
__global__ __launch_bounds__(256)
void edge_exp_kernel(const int* __restrict__ src, const int* __restrict__ dst,
                     int E, int nself, int H,
                     float* __restrict__ ebuf, const unsigned* __restrict__ mkey,
                     float* __restrict__ den)
{
    int idx = blockIdx.x * blockDim.x + threadIdx.x;
    int ET = E + nself;
    if (idx >= ET * H) return;
    int e = idx / H, h = idx % H;
    int d_ = (e < E) ? dst[e] : (e - E);
    float m = fdec(mkey[d_ * H + h]);
    float ex = expf(ebuf[idx] - m);
    ebuf[idx] = ex;
    atomicAdd(&den[d_ * H + h], ex);
}

// Pass 3: out[d] += h_src[s] * alpha   (block per edge, thread per channel)
__global__ __launch_bounds__(512)
void edge_agg_kernel(const int* __restrict__ src, const int* __restrict__ dst,
                     int E, int nself, int H, int C,
                     const float* __restrict__ hsrc, const float* __restrict__ ebuf,
                     const float* __restrict__ den, float* __restrict__ out)
{
    int e = blockIdx.x;
    int t = threadIdx.x;            // t in [0, H*C)
    int s_ = (e < E) ? src[e] : (e - E);
    int d_ = (e < E) ? dst[e] : (e - E);
    int h = t / C;
    float alpha = ebuf[e * H + h] / (den[d_ * H + h] + 1e-16f);
    atomicAdd(&out[(size_t)d_ * H * C + t], hsrc[(size_t)s_ * H * C + t] * alpha);
}

__global__ __launch_bounds__(256)
void addbias_kernel(float* __restrict__ x, const float* __restrict__ bias, int total, int D)
{
    int idx = blockIdx.x * blockDim.x + threadIdx.x;
    if (idx < total) x[idx] += bias[idx % D];
}

// x += bias + res; also emit bf16 copy
__global__ __launch_bounds__(256)
void add_res_cast_kernel(float* __restrict__ x, const float* __restrict__ bias,
                         const float* __restrict__ res, __hip_bfloat16* __restrict__ ob,
                         int total, int D)
{
    int idx = blockIdx.x * blockDim.x + threadIdx.x;
    if (idx >= total) return;
    float v = x[idx] + bias[idx % D] + res[idx];
    x[idx] = v;
    ob[idx] = __float2bfloat16(v);
}

// ---------------------------------------------------------------------------
// Host launch
// ---------------------------------------------------------------------------
extern "C" void kernel_launch(void* const* d_in, const int* in_sizes, int n_in,
                              void* d_out, int out_size, void* d_ws, size_t ws_size,
                              hipStream_t stream)
{
    (void)n_in; (void)out_size; (void)ws_size;
    const float* x_fine   = (const float*)d_in[0];
    const float* x_coarse = (const float*)d_in[1];
    const float* x_global = (const float*)d_in[2];
    const int*   ei_fc    = (const int*)d_in[3];
    const int*   ei_cg    = (const int*)d_in[4];
    const float* g_fine   = (const float*)d_in[5];
    const float* b_fine   = (const float*)d_in[6];
    const float* g_coarse = (const float*)d_in[7];
    const float* b_coarse = (const float*)d_in[8];
    const float* g_global = (const float*)d_in[9];
    const float* b_global = (const float*)d_in[10];
    const float* W_ip1    = (const float*)d_in[11];
    const float* b_ip1    = (const float*)d_in[12];
    const float* W_ip2    = (const float*)d_in[13];
    const float* b_ip2    = (const float*)d_in[14];
    const float* Wsrc1    = (const float*)d_in[15];
    const float* Wdst1    = (const float*)d_in[16];
    const float* asrc1    = (const float*)d_in[17];
    const float* adst1    = (const float*)d_in[18];
    const float* bias1    = (const float*)d_in[19];
    const float* Wsrc2    = (const float*)d_in[20];
    const float* Wdst2    = (const float*)d_in[21];
    const float* asrc2    = (const float*)d_in[22];
    const float* adst2    = (const float*)d_in[23];
    const float* bias2    = (const float*)d_in[24];
    const float* W_gp     = (const float*)d_in[25];
    const float* b_gp     = (const float*)d_in[26];
    const float* W_cp     = (const float*)d_in[27];
    const float* b_cp     = (const float*)d_in[28];
    const float* W_fp1    = (const float*)d_in[29];
    const float* b_fp1    = (const float*)d_in[30];
    const float* W_fp2    = (const float*)d_in[31];
    const float* b_fp2    = (const float*)d_in[32];

    const int E1 = in_sizes[3] / 2;   // 131072
    const int E2 = in_sizes[4] / 2;   // 4096
    float* out = (float*)d_out;

    // ---- workspace arena ---------------------------------------------------
    char* ws = (char*)d_ws;
    size_t off = 0;
    auto alloc = [&](size_t bytes) -> char* {
        char* p = ws + off;
        off += (bytes + 255) & ~(size_t)255;
        return p;
    };
    // region1: xf_ln bf16 [32768,1024] (64MB), reused as h_src1 f32 [32768,512]
    char* region1 = alloc(67108864);
    __hip_bfloat16* xf_ln  = (__hip_bfloat16*)region1;
    float*          h_src1 = (float*)region1;
    // region2: hid_f bf16 [32768,2048] (128MB), reused as fin_h bf16
    char* region2 = alloc(134217728);
    __hip_bfloat16* hid_f = (__hip_bfloat16*)region2;
    __hip_bfloat16* fin_h = (__hip_bfloat16*)region2;

    __hip_bfloat16* xf_bf   = (__hip_bfloat16*)alloc((size_t)N_FINE * OUT_P * 2);
    __hip_bfloat16* xc_ln   = (__hip_bfloat16*)alloc((size_t)N_COARSE * IN_F * 2);
    __hip_bfloat16* hid_c   = (__hip_bfloat16*)alloc((size_t)N_COARSE * HID * 2);
    __hip_bfloat16* xc0_bf  = (__hip_bfloat16*)alloc((size_t)N_COARSE * OUT_P * 2);
    float*          xg_ln_f = (float*)alloc((size_t)BSZ * CD * 4);
    __hip_bfloat16* xg_ln_b = (__hip_bfloat16*)alloc((size_t)BSZ * CD * 2);

    __hip_bfloat16* Wip1t  = (__hip_bfloat16*)alloc((size_t)IN_F * HID * 2);
    __hip_bfloat16* Wip2t  = (__hip_bfloat16*)alloc((size_t)HID * OUT_P * 2);
    __hip_bfloat16* Wsrc1t = (__hip_bfloat16*)alloc((size_t)OUT_P * CD * 2);
    __hip_bfloat16* Wdst1t = (__hip_bfloat16*)alloc((size_t)OUT_P * CD * 2);
    __hip_bfloat16* Wsrc2t = (__hip_bfloat16*)alloc((size_t)CD * CD * 2);
    __hip_bfloat16* Wdst2t = (__hip_bfloat16*)alloc((size_t)CD * CD * 2);
    __hip_bfloat16* Wgpt   = (__hip_bfloat16*)alloc((size_t)CD * LLM * 2);
    __hip_bfloat16* Wcpt   = (__hip_bfloat16*)alloc((size_t)CD * LLM * 2);
    __hip_bfloat16* Wfp1t  = (__hip_bfloat16*)alloc((size_t)OUT_P * HID * 2);
    __hip_bfloat16* Wfp2t  = (__hip_bfloat16*)alloc((size_t)HID * LLM * 2);

    float*    h_dst1 = (float*)alloc((size_t)N_COARSE * CD * 4);
    float*    ssrc1  = (float*)alloc((size_t)N_FINE * H1 * 4);
    float*    sdst1  = (float*)alloc((size_t)N_COARSE * H1 * 4);
    unsigned* mkey1  = (unsigned*)alloc((size_t)N_COARSE * H1 * 4);
    float*    den1   = (float*)alloc((size_t)N_COARSE * H1 * 4);
    float*    ebuf1  = (float*)alloc((size_t)(E1 + N_COARSE) * H1 * 4);
    float*    xc1    = (float*)alloc((size_t)N_COARSE * CD * 4);
    __hip_bfloat16* xc2_bf = (__hip_bfloat16*)alloc((size_t)N_COARSE * CD * 2);
    float*    h_src2 = (float*)alloc((size_t)N_COARSE * CD * 4);
    float*    h_dst2 = (float*)alloc((size_t)BSZ * CD * 4);
    float*    ssrc2  = (float*)alloc((size_t)N_COARSE * H2 * 4);
    float*    sdst2  = (float*)alloc((size_t)BSZ * H2 * 4);
    unsigned* mkey2  = (unsigned*)alloc((size_t)BSZ * H2 * 4);
    float*    den2   = (float*)alloc((size_t)BSZ * H2 * 4);
    float*    ebuf2  = (float*)alloc((size_t)(E2 + BSZ) * H2 * 4);
    float*    xg2    = (float*)alloc((size_t)BSZ * CD * 4);
    __hip_bfloat16* xg2_bf = (__hip_bfloat16*)alloc((size_t)BSZ * CD * 2);

    const long long OBS = (long long)OUT_ROWS * LLM;   // per-batch output stride
    auto cdiv = [](int a, int b) { return (a + b - 1) / b; };

    // ---- weight casts (f32 [K,N] -> bf16 [N,K]) ----------------------------
    auto castT = [&](const float* W, __hip_bfloat16* Wt, int K, int N) {
        castT_kernel<<<cdiv(K * N, 256), 256, 0, stream>>>(W, Wt, K, N);
    };
    castT(W_ip1, Wip1t, IN_F, HID);
    castT(W_ip2, Wip2t, HID, OUT_P);
    castT(Wsrc1, Wsrc1t, OUT_P, CD);
    castT(Wdst1, Wdst1t, OUT_P, CD);
    castT(Wsrc2, Wsrc2t, CD, CD);
    castT(Wdst2, Wdst2t, CD, CD);
    castT(W_gp,  Wgpt,  CD, LLM);
    castT(W_cp,  Wcpt,  CD, LLM);
    castT(W_fp1, Wfp1t, OUT_P, HID);
    castT(W_fp2, Wfp2t, HID, LLM);

    // ---- layer norms -------------------------------------------------------
    ln_kernel<<<N_FINE,   256, 0, stream>>>(x_fine,   g_fine,   b_fine,   nullptr, xf_ln, IN_F, 0);
    ln_kernel<<<N_COARSE, 256, 0, stream>>>(x_coarse, g_fine,   b_fine,   nullptr, xc_ln, IN_F, 0);
    ln_kernel<<<BSZ,      256, 0, stream>>>(x_global, g_global, b_global, xg_ln_f, xg_ln_b, CD, 0);

    // ---- GEMM helper (wave tile 32x64, block tile 128x64) ------------------
    auto gemm = [&](const __hip_bfloat16* A, const __hip_bfloat16* Bt, const float* bias,
                    float* oF, __hip_bfloat16* oB, int M, int N, int K,
                    int grows, int goff, int rstride, int relu) {
        dim3 grid(N / 64, cdiv(M, 128));
        gemm_wmma_bf16_kernel<<<grid, 128, 0, stream>>>(A, Bt, bias, oF, oB,
                                                        M, N, K, grows, goff, OBS, rstride, relu);
    };

    // fine & coarse input MLPs
    gemm(xf_ln, Wip1t, b_ip1, nullptr, hid_f,  N_FINE,   HID,   IN_F, 0, 0, HID,   1);
    gemm(hid_f, Wip2t, b_ip2, nullptr, xf_bf,  N_FINE,   OUT_P, HID,  0, 0, OUT_P, 0);
    gemm(xc_ln, Wip1t, b_ip1, nullptr, hid_c,  N_COARSE, HID,   IN_F, 0, 0, HID,   1);
    gemm(hid_c, Wip2t, b_ip2, nullptr, xc0_bf, N_COARSE, OUT_P, HID,  0, 0, OUT_P, 0);

    // GAT1 projections (region1 now reused as h_src1)
    gemm(xf_bf,  Wsrc1t, nullptr, h_src1, nullptr, N_FINE,   CD, OUT_P, 0, 0, CD, 0);
    gemm(xc0_bf, Wdst1t, nullptr, h_dst1, nullptr, N_COARSE, CD, OUT_P, 0, 0, CD, 0);

    // GAT1 softmax + aggregate
    scores_kernel<<<cdiv(N_FINE * H1, 256),   256, 0, stream>>>(h_src1, asrc1, ssrc1, N_FINE,   H1, ATT);
    scores_kernel<<<cdiv(N_COARSE * H1, 256), 256, 0, stream>>>(h_dst1, adst1, sdst1, N_COARSE, H1, ATT);
    fill_u32_kernel<<<cdiv(N_COARSE * H1, 256), 256, 0, stream>>>(mkey1, 0x007FFFFFu, N_COARSE * H1);
    hipMemsetAsync(den1, 0, (size_t)N_COARSE * H1 * 4, stream);
    hipMemsetAsync(xc1,  0, (size_t)N_COARSE * CD * 4, stream);
    {
        const int ET = E1 + N_COARSE;
        edge_max_kernel<<<cdiv(ET * H1, 256), 256, 0, stream>>>(ei_fc, ei_fc + E1, E1, N_COARSE, H1,
                                                                ssrc1, sdst1, ebuf1, mkey1);
        edge_exp_kernel<<<cdiv(ET * H1, 256), 256, 0, stream>>>(ei_fc, ei_fc + E1, E1, N_COARSE, H1,
                                                                ebuf1, mkey1, den1);
        edge_agg_kernel<<<ET, H1 * ATT, 0, stream>>>(ei_fc, ei_fc + E1, E1, N_COARSE, H1, ATT,
                                                     h_src1, ebuf1, den1, xc1);
    }
    addbias_kernel<<<cdiv(N_COARSE * CD, 256), 256, 0, stream>>>(xc1, bias1, N_COARSE * CD, CD);
    ln_kernel<<<N_COARSE, 256, 0, stream>>>(xc1, g_coarse, b_coarse, nullptr, xc2_bf, CD, 1);

    // GAT2 projections
    gemm(xc2_bf,  Wsrc2t, nullptr, h_src2, nullptr, N_COARSE, CD, CD, 0, 0, CD, 0);
    gemm(xg_ln_b, Wdst2t, nullptr, h_dst2, nullptr, BSZ,      CD, CD, 0, 0, CD, 0);

    // GAT2 softmax + aggregate
    scores_kernel<<<cdiv(N_COARSE * H2, 256), 256, 0, stream>>>(h_src2, asrc2, ssrc2, N_COARSE, H2, OUT_P);
    scores_kernel<<<cdiv(BSZ * H2, 256),      256, 0, stream>>>(h_dst2, adst2, sdst2, BSZ,      H2, OUT_P);
    fill_u32_kernel<<<cdiv(BSZ * H2, 256), 256, 0, stream>>>(mkey2, 0x007FFFFFu, BSZ * H2);
    hipMemsetAsync(den2, 0, (size_t)BSZ * H2 * 4, stream);
    hipMemsetAsync(xg2,  0, (size_t)BSZ * CD * 4, stream);
    {
        const int ET = E2 + BSZ;
        edge_max_kernel<<<cdiv(ET * H2, 256), 256, 0, stream>>>(ei_cg, ei_cg + E2, E2, BSZ, H2,
                                                                ssrc2, sdst2, ebuf2, mkey2);
        edge_exp_kernel<<<cdiv(ET * H2, 256), 256, 0, stream>>>(ei_cg, ei_cg + E2, E2, BSZ, H2,
                                                                ebuf2, mkey2, den2);
        edge_agg_kernel<<<ET, H2 * OUT_P, 0, stream>>>(ei_cg, ei_cg + E2, E2, BSZ, H2, OUT_P,
                                                       h_src2, ebuf2, den2, xg2);
    }
    add_res_cast_kernel<<<cdiv(BSZ * CD, 256), 256, 0, stream>>>(xg2, bias2, xg_ln_f, xg2_bf,
                                                                 BSZ * CD, CD);

    // ---- output heads (mapped into [B, 1089, 4096]) ------------------------
    gemm(xg2_bf, Wgpt, b_gp, out, nullptr, BSZ,      LLM, CD, 1,    0,  LLM, 0);
    gemm(xc2_bf, Wcpt, b_cp, out, nullptr, N_COARSE, LLM, CD, 64,   1,  LLM, 0);
    gemm(xf_bf,  Wfp1t, b_fp1, nullptr, fin_h, N_FINE, HID, OUT_P, 0, 0, HID, 1);
    gemm(fin_h,  Wfp2t, b_fp2, out, nullptr, N_FINE,  LLM, HID, 1024, 65, LLM, 0);
}
